// PatchedPhi3Attention_59897613910436
// MI455X (gfx1250) — matrix operand order repaired and checked by reading, not compiled, not run on hardware
//
#include <hip/hip_runtime.h>
#include <hip/hip_bf16.h>
#include <math.h>

// ---------------------------------------------------------------------------
// Problem constants (from reference): B=2, S=2048, H=32, HKV=8, D=128, BS=64
// ---------------------------------------------------------------------------
constexpr int BB   = 2;
constexpr int SS   = 2048;
constexpr int HH   = 32;
constexpr int HKV  = 8;
constexpr int DD   = 128;
constexpr int HID  = HH * DD;              // 4096
constexpr int NQKV = (HH + 2 * HKV) * DD;  // 6144
constexpr int TT   = BB * SS;              // 4096

typedef __attribute__((ext_vector_type(8)))  float  v8f;
typedef __attribute__((ext_vector_type(4)))  __bf16 v4bf;
typedef __attribute__((ext_vector_type(8)))  __bf16 v8bf;
typedef __attribute__((ext_vector_type(16))) __bf16 v16bf;

union bfu16 { v16bf v; v8bf h[2]; };

struct bf2 { __bf16 hi, lo; };

// x = (float)hi + (float)lo to ~16 mantissa bits
__device__ __forceinline__ bf2 split_bf16(float x) {
    bf2 r;
    r.hi = (__bf16)x;
    r.lo = (__bf16)(x - (float)r.hi);
    return r;
}

// split a float4 into hi/lo 4-wide bf16 vectors
__device__ __forceinline__ void split4(const float4 v, v4bf& h4, v4bf& l4) {
    bf2 a = split_bf16(v.x), b = split_bf16(v.y), c = split_bf16(v.z), d = split_bf16(v.w);
    h4[0] = a.hi; h4[1] = b.hi; h4[2] = c.hi; h4[3] = d.hi;
    l4[0] = a.lo; l4[1] = b.lo; l4[2] = c.lo; l4[3] = d.lo;
}

__device__ __forceinline__ v16bf ld_frag(const __bf16* p0, const __bf16* p1) {
    bfu16 u;
    u.h[0] = *(const v8bf*)p0;
    u.h[1] = *(const v8bf*)p1;
    return u.v;
}

#define WMMA_BF16(A, B, C) \
    __builtin_amdgcn_wmma_f32_16x16x32_bf16(false, (A), false, (B), (short)0, (C), false, false)

// ---------------------------------------------------------------------------
// Split-bf16 (bf16x3) WMMA GEMM:  C[M,N] = A[M,K] * W[N,K]^T, fp32 accumulate.
// Block tile 128x64, K-chunk 32, 256 threads = 8 waves (4m x 2n), each wave a
// 32x32 tile = 2x2 WMMA 16x16x32 tiles, 3 WMMAs per tile per chunk.
// LDS rows hold [0..31]=hi plane, [32..63]=lo plane, pad to 72 bf16 (144B).
// ---------------------------------------------------------------------------
__global__ __launch_bounds__(256) void gemm_bf16x3_wmma_kernel(
    const float* __restrict__ A, const float* __restrict__ W,
    float* __restrict__ C, int M, int N, int K)
{
    __shared__ __bf16 As[2][128][72];
    __shared__ __bf16 Bs[2][64][72];

    const int tid  = threadIdx.x;
    const int lane = tid & 31;
    const int wave = tid >> 5;
    const int wm   = wave & 3;
    const int wn   = wave >> 2;
    const int l15  = lane & 15;
    const int lh   = lane >> 4;

    const int rowBase = blockIdx.y * 128;
    const int colBase = blockIdx.x * 64;

    const int sm = tid >> 3;        // staging row group 0..31
    const int sk = (tid & 7) << 2;  // staging k offset 0,4,...,28

    float4 aR[4];
    float4 bR[2];

    // ---- stage chunk 0 ----
#pragma unroll
    for (int j = 0; j < 4; ++j)
        aR[j] = *(const float4*)(A + (size_t)(rowBase + sm + 32 * j) * K + sk);
#pragma unroll
    for (int j = 0; j < 2; ++j)
        bR[j] = *(const float4*)(W + (size_t)(colBase + sm + 32 * j) * K + sk);
#pragma unroll
    for (int j = 0; j < 4; ++j) {
        v4bf h4, l4;
        split4(aR[j], h4, l4);
        *(v4bf*)&As[0][sm + 32 * j][sk]      = h4;
        *(v4bf*)&As[0][sm + 32 * j][32 + sk] = l4;
    }
#pragma unroll
    for (int j = 0; j < 2; ++j) {
        v4bf h4, l4;
        split4(bR[j], h4, l4);
        *(v4bf*)&Bs[0][sm + 32 * j][sk]      = h4;
        *(v4bf*)&Bs[0][sm + 32 * j][32 + sk] = l4;
    }
    __syncthreads();

    v8f acc00 = {}, acc01 = {}, acc10 = {}, acc11 = {};

    const int nk = K >> 5;
    int buf = 0;
    for (int kt = 0; kt < nk; ++kt) {
        const bool more = (kt + 1) < nk;
        if (more) {
            const int kn = (kt + 1) << 5;
#pragma unroll
            for (int j = 0; j < 4; ++j)
                aR[j] = *(const float4*)(A + (size_t)(rowBase + sm + 32 * j) * K + kn + sk);
#pragma unroll
            for (int j = 0; j < 2; ++j)
                bR[j] = *(const float4*)(W + (size_t)(colBase + sm + 32 * j) * K + kn + sk);
        }

        // A fragments: contiguous K run at k = lh*16
        {
            const int ka0 = lh * 16, ka1 = lh * 16 + 8;
            const int ra0 = wm * 32 + l15, ra1 = ra0 + 16;
            v16bf ah0 = ld_frag(&As[buf][ra0][ka0],      &As[buf][ra0][ka1]);
            v16bf al0 = ld_frag(&As[buf][ra0][32 + ka0], &As[buf][ra0][32 + ka1]);
            v16bf ah1 = ld_frag(&As[buf][ra1][ka0],      &As[buf][ra1][ka1]);
            v16bf al1 = ld_frag(&As[buf][ra1][32 + ka0], &As[buf][ra1][32 + ka1]);
            // B fragments: contiguous K run at k = lh*16
            const int kb0 = lh * 16;
            const int rb0 = wn * 32 + l15, rb1 = rb0 + 16;
            v16bf bh0 = ld_frag(&Bs[buf][rb0][kb0],      &Bs[buf][rb0][kb0 + 8]);
            v16bf bl0 = ld_frag(&Bs[buf][rb0][32 + kb0], &Bs[buf][rb0][32 + kb0 + 8]);
            v16bf bh1 = ld_frag(&Bs[buf][rb1][kb0],      &Bs[buf][rb1][kb0 + 8]);
            v16bf bl1 = ld_frag(&Bs[buf][rb1][32 + kb0], &Bs[buf][rb1][32 + kb0 + 8]);

            acc00 = WMMA_BF16(ah0, bh0, acc00);
            acc00 = WMMA_BF16(ah0, bl0, acc00);
            acc00 = WMMA_BF16(al0, bh0, acc00);

            acc01 = WMMA_BF16(ah0, bh1, acc01);
            acc01 = WMMA_BF16(ah0, bl1, acc01);
            acc01 = WMMA_BF16(al0, bh1, acc01);

            acc10 = WMMA_BF16(ah1, bh0, acc10);
            acc10 = WMMA_BF16(ah1, bl0, acc10);
            acc10 = WMMA_BF16(al1, bh0, acc10);

            acc11 = WMMA_BF16(ah1, bh1, acc11);
            acc11 = WMMA_BF16(ah1, bl1, acc11);
            acc11 = WMMA_BF16(al1, bh1, acc11);
        }

        if (more) {
#pragma unroll
            for (int j = 0; j < 4; ++j) {
                v4bf h4, l4;
                split4(aR[j], h4, l4);
                *(v4bf*)&As[buf ^ 1][sm + 32 * j][sk]      = h4;
                *(v4bf*)&As[buf ^ 1][sm + 32 * j][32 + sk] = l4;
            }
#pragma unroll
            for (int j = 0; j < 2; ++j) {
                v4bf h4, l4;
                split4(bR[j], h4, l4);
                *(v4bf*)&Bs[buf ^ 1][sm + 32 * j][sk]      = h4;
                *(v4bf*)&Bs[buf ^ 1][sm + 32 * j][32 + sk] = l4;
            }
            __syncthreads();
            buf ^= 1;
        }
    }

    // C layout: VGPR i -> row (lh*8 + i), col = l15
#pragma unroll
    for (int i = 0; i < 8; ++i) {
        const int row = rowBase + wm * 32 + lh * 8 + i;
        const int col = colBase + wn * 32 + l15;
        C[(size_t)row * N + col]             = acc00[i];
        C[(size_t)row * N + col + 16]        = acc01[i];
        C[(size_t)(row + 16) * N + col]      = acc10[i];
        C[(size_t)(row + 16) * N + col + 16] = acc11[i];
    }
}

// ---------------------------------------------------------------------------
// RoPE (in place for Q) + K(roped)/V reorg into (b, hkv, s, d) fp32 buffers.
// ---------------------------------------------------------------------------
__global__ __launch_bounds__(256) void rope_scatter_kernel(
    float* __restrict__ qkv, float* __restrict__ kbuf, float* __restrict__ vbuf,
    const int* __restrict__ pos)
{
    const int idx = blockIdx.x * 256 + threadIdx.x;
    const int d    = idx & 63;
    const int rest = idx >> 6;
    const int hh   = rest % 48;
    const int t    = rest / 48;
    if (t >= TT) return;

    const int b = t / SS, srow = t % SS;
    if (hh < 40) {
        const float p  = (float)pos[t];
        const float fr = p * __powf(10000.f, -(float)(2 * d) / 128.f);
        const float c  = __cosf(fr);
        const float s  = __sinf(fr);
        if (hh < 32) {
            float* x = qkv + (size_t)t * NQKV + hh * DD;
            const float x1 = x[d], x2 = x[d + 64];
            x[d]      = x1 * c - x2 * s;
            x[d + 64] = x2 * c + x1 * s;
        } else {
            const int kh = hh - 32;
            const float* x = qkv + (size_t)t * NQKV + HID + kh * DD;
            const float x1 = x[d], x2 = x[d + 64];
            float* o = kbuf + (((size_t)(b * HKV + kh) * SS + srow) << 7);
            o[d]      = x1 * c - x2 * s;
            o[d + 64] = x2 * c + x1 * s;
        }
    } else {
        const int vh = hh - 40;
        const float* x = qkv + (size_t)t * NQKV + HID + HKV * DD + vh * DD;
        float* o = vbuf + (((size_t)(b * HKV + vh) * SS + srow) << 7);
        o[d]      = x[d];
        o[d + 64] = x[d + 64];
    }
}

// ---------------------------------------------------------------------------
// Flash-style sliding-window causal GQA attention, split-bf16 WMMA, 32-key
// chunks. Grid (S/64, H, B); 128 threads = 4 waves; wave owns 16 q-rows.
//   QK^T: 2 key-groups x 4 d-chunks x 3 splits = 24 WMMAs / chunk / wave
//   P*V : 8 d-tiles x 3 splits                = 24 WMMAs / chunk / wave
// ---------------------------------------------------------------------------
__global__ __launch_bounds__(128) void attn_wmma_bf16_kernel(
    const float* __restrict__ qkv, const float* __restrict__ kbuf,
    const float* __restrict__ vbuf, float* __restrict__ obuf,
    const int* __restrict__ winp)
{
    __shared__ __bf16 Ks[32][264];    // [key][0..127 hi | 128..255 lo | pad]
    __shared__ __bf16 Vt[128][72];    // [d][0..31 hi(key) | 32..63 lo(key) | pad]
    __shared__ __bf16 Ps[4][16][72];  // [wave][qrow][0..31 hi | 32..63 lo | pad]

    const int Wwin = winp[0];
    const int b   = blockIdx.z;
    const int h   = blockIdx.y;
    const int qb  = blockIdx.x << 6;
    const int hkv = h >> 2;           // G = 4
    const int tid = threadIdx.x;
    const int wv  = tid >> 5;
    const int lane = tid & 31;
    const int l15 = lane & 15;
    const int lh  = lane >> 4;
    const float scale = 0.08838834764831845f;  // 1/sqrt(128)

    // Q tile (16 x 128) as hi/lo A-fragments, scale folded in before split
    v16bf qh[4], ql[4];
    {
        const float* qp = qkv + (size_t)(b * SS + qb + wv * 16 + l15) * NQKV + h * DD;
#pragma unroll
        for (int dc = 0; dc < 4; ++dc) {
            bfu16 H, L;
#pragma unroll
            for (int r = 0; r < 2; ++r) {
                const int dbase = dc * 32 + lh * 16 + r * 8;
#pragma unroll
                for (int e = 0; e < 8; ++e) {
                    const bf2 t = split_bf16(qp[dbase + e] * scale);
                    ((__bf16*)&H.v)[r * 8 + e] = t.hi;
                    ((__bf16*)&L.v)[r * 8 + e] = t.lo;
                }
            }
            qh[dc] = H.v;
            ql[dc] = L.v;
        }
    }

    float mrun[8], lrun[8];
    v8f oacc[8] = {};
#pragma unroll
    for (int i = 0; i < 8; ++i) { mrun[i] = -1e30f; lrun[i] = 0.f; }

    int klo = qb - (Wwin - 1);
    if (klo < 0) klo = 0;
    klo &= ~31;
    const size_t kvbase = ((size_t)(b * HKV + hkv) * SS) << 7;

    for (int kc = klo; kc <= qb + 63; kc += 32) {
        __syncthreads();
        // cooperative 32x128 K and V chunk load + bf16 split (V transposed)
#pragma unroll
        for (int j = 0; j < 8; ++j) {
            const int idx = tid + j * 128;
            const int key = idx >> 5;
            const int d   = (idx & 31) << 2;
            const float4 kx = *(const float4*)(kbuf + kvbase + ((size_t)(kc + key) << 7) + d);
            v4bf h4, l4;
            split4(kx, h4, l4);
            *(v4bf*)&Ks[key][d]       = h4;
            *(v4bf*)&Ks[key][128 + d] = l4;

            const float4 vx = *(const float4*)(vbuf + kvbase + ((size_t)(kc + key) << 7) + d);
            bf2 t;
            t = split_bf16(vx.x); Vt[d + 0][key] = t.hi; Vt[d + 0][32 + key] = t.lo;
            t = split_bf16(vx.y); Vt[d + 1][key] = t.hi; Vt[d + 1][32 + key] = t.lo;
            t = split_bf16(vx.z); Vt[d + 2][key] = t.hi; Vt[d + 2][32 + key] = t.lo;
            t = split_bf16(vx.w); Vt[d + 3][key] = t.hi; Vt[d + 3][32 + key] = t.lo;
        }
        __syncthreads();

        // scores: Q (16x128) x K^T (128 x 32) as two 16x16 tiles
        v8f sc0 = {}, sc1 = {};
#pragma unroll
        for (int dc = 0; dc < 4; ++dc) {
            const int doff = dc * 32 + lh * 16;
            v16bf kh0 = ld_frag(&Ks[l15][doff],            &Ks[l15][doff + 8]);
            v16bf kl0 = ld_frag(&Ks[l15][128 + doff],      &Ks[l15][128 + doff + 8]);
            v16bf kh1 = ld_frag(&Ks[16 + l15][doff],       &Ks[16 + l15][doff + 8]);
            v16bf kl1 = ld_frag(&Ks[16 + l15][128 + doff], &Ks[16 + l15][128 + doff + 8]);
            sc0 = WMMA_BF16(qh[dc], kh0, sc0);
            sc0 = WMMA_BF16(qh[dc], kl0, sc0);
            sc0 = WMMA_BF16(ql[dc], kh0, sc0);
            sc1 = WMMA_BF16(qh[dc], kh1, sc1);
            sc1 = WMMA_BF16(qh[dc], kl1, sc1);
            sc1 = WMMA_BF16(ql[dc], kh1, sc1);
        }

        // online softmax over 32 keys (two 16-wide groups)
        const int kj0 = kc + l15;
        const int kj1 = kc + 16 + l15;
#pragma unroll
        for (int i = 0; i < 8; ++i) {
            const int qi = qb + wv * 16 + lh * 8 + i;
            const bool v0 = (kj0 <= qi) && ((qi - kj0) < Wwin);
            const bool v1 = (kj1 <= qi) && ((qi - kj1) < Wwin);
            const float s0 = v0 ? sc0[i] : -1e30f;
            const float s1 = v1 ? sc1[i] : -1e30f;
            float rm = fmaxf(s0, s1);
            rm = fmaxf(rm, __shfl_xor(rm, 1, 16));
            rm = fmaxf(rm, __shfl_xor(rm, 2, 16));
            rm = fmaxf(rm, __shfl_xor(rm, 4, 16));
            rm = fmaxf(rm, __shfl_xor(rm, 8, 16));
            const float mnew  = fmaxf(mrun[i], rm);
            const float alpha = __expf(mrun[i] - mnew);
            const float p0 = v0 ? __expf(s0 - mnew) : 0.f;
            const float p1 = v1 ? __expf(s1 - mnew) : 0.f;
            float ps = p0 + p1;
            ps += __shfl_xor(ps, 1, 16);
            ps += __shfl_xor(ps, 2, 16);
            ps += __shfl_xor(ps, 4, 16);
            ps += __shfl_xor(ps, 8, 16);
            lrun[i] = lrun[i] * alpha + ps;
            mrun[i] = mnew;
#pragma unroll
            for (int nt = 0; nt < 8; ++nt) oacc[nt][i] *= alpha;

            // P -> LDS as hi/lo bf16 (row = qrow, col = key)
            const int prow = lh * 8 + i;
            bf2 t;
            t = split_bf16(p0);
            Ps[wv][prow][l15]      = t.hi;
            Ps[wv][prow][32 + l15] = t.lo;
            t = split_bf16(p1);
            Ps[wv][prow][16 + l15] = t.hi;
            Ps[wv][prow][48 + l15] = t.lo;
        }
        // wave-private LDS round trip: wait for ds stores, block compiler reorder
        asm volatile("s_wait_dscnt 0" ::: "memory");

        // P as 16x32 A-fragments (hi/lo)
        const int ka0 = lh * 16, ka1 = lh * 16 + 8;
        v16bf ph = ld_frag(&Ps[wv][l15][ka0],      &Ps[wv][l15][ka1]);
        v16bf pl = ld_frag(&Ps[wv][l15][32 + ka0], &Ps[wv][l15][32 + ka1]);

        // O += P (16x32) x V (32x128)
#pragma unroll
        for (int nt = 0; nt < 8; ++nt) {
            const int d = nt * 16 + l15;
            v16bf vh = ld_frag(&Vt[d][lh * 16],      &Vt[d][lh * 16 + 8]);
            v16bf vl = ld_frag(&Vt[d][32 + lh * 16], &Vt[d][32 + lh * 16 + 8]);
            oacc[nt] = WMMA_BF16(ph, vh, oacc[nt]);
            oacc[nt] = WMMA_BF16(ph, vl, oacc[nt]);
            oacc[nt] = WMMA_BF16(pl, vh, oacc[nt]);
        }
    }

    float inv[8];
#pragma unroll
    for (int i = 0; i < 8; ++i) inv[i] = 1.f / lrun[i];

    float* op = obuf + (size_t)(b * SS + qb + wv * 16) * HID + h * DD;
#pragma unroll
    for (int nt = 0; nt < 8; ++nt)
#pragma unroll
        for (int i = 0; i < 8; ++i)
            op[(size_t)(lh * 8 + i) * HID + nt * 16 + l15] = oacc[nt][i] * inv[i];
}

// ---------------------------------------------------------------------------
// Launcher. Workspace layout (fp32):
//   qkv  : T*6144  (96 MB) | kbuf/vbuf : B*HKV*S*D (16 MB each) | obuf : T*4096
// ---------------------------------------------------------------------------
extern "C" void kernel_launch(void* const* d_in, const int* in_sizes, int n_in,
                              void* d_out, int out_size, void* d_ws, size_t ws_size,
                              hipStream_t stream) {
    const float* hidden = (const float*)d_in[0];
    const float* w_qkv  = (const float*)d_in[1];
    const float* w_o    = (const float*)d_in[2];
    const int*   pos    = (const int*)d_in[5];
    const int*   win    = (const int*)d_in[7];
    float* out = (float*)d_out;

    float* qkvb = (float*)d_ws;
    float* kb   = qkvb + (size_t)TT * NQKV;
    float* vb   = kb + (size_t)BB * HKV * SS * DD;
    float* ob   = vb + (size_t)BB * HKV * SS * DD;

    (void)in_sizes; (void)n_in; (void)out_size; (void)ws_size;

    // 1) QKV projection: (T x HID) * (NQKV x HID)^T
    gemm_bf16x3_wmma_kernel<<<dim3(NQKV / 64, TT / 128), 256, 0, stream>>>(
        hidden, w_qkv, qkvb, TT, NQKV, HID);

    // 2) RoPE (q in place) + K/V reorg
    rope_scatter_kernel<<<(TT * 48 * 64) / 256, 256, 0, stream>>>(qkvb, kb, vb, pos);

    // 3) sliding-window causal GQA attention
    attn_wmma_bf16_kernel<<<dim3(SS / 64, HH, BB), 128, 0, stream>>>(qkvb, kb, vb, ob, win);

    // 4) output projection: (T x HID) * (HID x HID)^T
    gemm_bf16x3_wmma_kernel<<<dim3(HID / 64, TT / 128), 256, 0, stream>>>(
        ob, w_o, out, TT, HID, HID);
}
